// LiquidLayer_63531156242850
// MI455X (gfx1250) — compile-verified, matrix operand order checked by reading, not run on hardware
//
#include <hip/hip_runtime.h>
#include <hip/hip_bf16.h>
#include <math.h>

// ---------------------------------------------------------------------------
// Problem constants (from the reference)
// ---------------------------------------------------------------------------
#define BATCH   4
#define TLEN    2048
#define HID     2048
#define NRES    4
#define RSIZE   256
#define DDIM    (NRES * RSIZE)   // 1024
#define MROWS   (BATCH * TLEN)   // 8192
#define LEAKC   0.1f
#define EPSC    1e-6f

typedef __bf16 v16bf __attribute__((ext_vector_type(16)));
typedef float  v8f   __attribute__((ext_vector_type(8)));

#define WMMA_BF16(A, B, C) \
    __builtin_amdgcn_wmma_f32_16x16x32_bf16(false, (A), false, (B), (short)0, (C), false, false)

// f32 -> bf16 RNE, result in low 16 bits
static __device__ __forceinline__ unsigned bf16_rne(float f) {
    unsigned x = __float_as_uint(f);
    return (x + 0x7FFFu + ((x >> 16) & 1u)) >> 16;
}
// 2-term split: f ~= hi + lo (each bf16)
static __device__ __forceinline__ void split2(float f, unsigned& hb, unsigned& lb) {
    hb = bf16_rne(f);
    float hf = __uint_as_float(hb << 16);
    lb = bf16_rne(f - hf);
}

// ---------------------------------------------------------------------------
// Fragment-native pack layout (shared by A and B operands):
//   record = tile16 * (K/32) + kchunk          (tile16: M/16 rows or N/16 cols)
//   per record: 32 lanes x 16 dwords (64 B/lane, hi[0..7] then lo[0..7])
//   dword d of lane (p = lane&15, half = lane>>4) holds K-pair:
//     d<4 : kk = kchunk*32 + half*8 + 2d
//     d>=4: kk = kchunk*32 + 16 + half*8 + 2(d-4)
//   -> a_hi/a_lo v16bf fragment is ONE contiguous 32-byte load per lane.
// ---------------------------------------------------------------------------

// Pack a row-major [Rows, K] f32 matrix (lane dim = rows). One thread per hi-dword.
__global__ __launch_bounds__(256)
void pack_split_rows(const float* __restrict__ src, unsigned* __restrict__ dst,
                     int K, int kc_log2, int total) {
    const int idx = blockIdx.x * 256 + threadIdx.x;
    if (idx >= total) return;
    const int rec  = idx >> 8;
    const int rem  = idx & 255;
    const int lane = rem >> 3;
    const int d    = rem & 7;
    const int tile = rec >> kc_log2;
    const int kch  = rec & ((1 << kc_log2) - 1);
    const int m    = tile * 16 + (lane & 15);
    const int half = lane >> 4;
    const int kk   = (kch << 5) + ((d < 4) ? (half * 8 + 2 * d)
                                           : (16 + half * 8 + 2 * (d - 4)));
    const float2 v = *reinterpret_cast<const float2*>(src + (size_t)m * K + kk);
    unsigned h0, l0, h1, l1;
    split2(v.x, h0, l0);
    split2(v.y, h1, l1);
    const size_t base = ((size_t)rec << 9) + (size_t)lane * 16 + d;
    dst[base]     = h0 | (h1 << 16);
    dst[base + 8] = l0 | (l1 << 16);
}

// Pack a row-major [K, N] f32 matrix by columns (lane dim = cols).
__global__ __launch_bounds__(256)
void pack_split_cols(const float* __restrict__ src, unsigned* __restrict__ dst,
                     int N, int kc_log2, int total) {
    const int idx = blockIdx.x * 256 + threadIdx.x;
    if (idx >= total) return;
    const int rec  = idx >> 8;
    const int rem  = idx & 255;
    const int lane = rem >> 3;
    const int d    = rem & 7;
    const int tile = rec >> kc_log2;
    const int kch  = rec & ((1 << kc_log2) - 1);
    const int n    = tile * 16 + (lane & 15);
    const int half = lane >> 4;
    const int kk   = (kch << 5) + ((d < 4) ? (half * 8 + 2 * d)
                                           : (16 + half * 8 + 2 * (d - 4)));
    const float f0 = src[(size_t)kk * N + n];
    const float f1 = src[(size_t)(kk + 1) * N + n];
    unsigned h0, l0, h1, l1;
    split2(f0, h0, l0);
    split2(f1, h1, l1);
    const size_t base = ((size_t)rec << 9) + (size_t)lane * 16 + d;
    dst[base]     = h0 | (h1 << 16);
    dst[base + 8] = l0 | (l1 << 16);
}

// ---------------------------------------------------------------------------
// GEMM on pre-packed operands. C[m,n] = sum_k A[m,k]*B[k,n] + bias[n] (+resid)
// via bf16-split WMMA: hi*hi + hi*lo + lo*hi (~f32 accuracy).
// Block: 128 threads (4 waves). Tile: 16 (M) x 512 (N); wave owns 8 N-subtiles.
// Inner loop: 18 vector loads + 24 v_wmma, 2 pointer bumps. No LDS, ~no VALU.
// grid = (N/512, M/16). KC = K/32.
// ---------------------------------------------------------------------------
__global__ __launch_bounds__(128)
void liquid_gemm_wmma(const unsigned* __restrict__ Ap,
                      const unsigned* __restrict__ Bp,
                      const float* __restrict__ bias,
                      const float* __restrict__ resid,
                      float* __restrict__ C,
                      int M, int N, int KC) {
    const int tid  = threadIdx.x;
    const int wave = tid >> 5;
    const int lane = tid & 31;
    const int half = lane >> 4;
    const int l16  = lane & 15;
    const int m0    = blockIdx.y * 16;
    const int nbase = blockIdx.x * 512 + wave * 128;

    const unsigned* aptr = Ap + (((size_t)blockIdx.y * KC) << 9) + (size_t)lane * 16;
    const unsigned* bptr = Bp + (((size_t)(nbase >> 4) * KC) << 9) + (size_t)lane * 16;
    const size_t tstride = (size_t)KC << 9;   // dwords per N-tile (all K-chunks)

    v8f acc[8];
    const v8f vzero = {0.f, 0.f, 0.f, 0.f, 0.f, 0.f, 0.f, 0.f};
#pragma unroll
    for (int t = 0; t < 8; ++t) acc[t] = vzero;

    for (int kc = 0; kc < KC; ++kc) {
        const v16bf a_hi = *reinterpret_cast<const v16bf*>(aptr);
        const v16bf a_lo = *reinterpret_cast<const v16bf*>(aptr + 8);
#pragma unroll
        for (int t = 0; t < 8; ++t) {
            const unsigned* bp = bptr + (size_t)t * tstride;
            const v16bf b_hi = *reinterpret_cast<const v16bf*>(bp);
            const v16bf b_lo = *reinterpret_cast<const v16bf*>(bp + 8);
            acc[t] = WMMA_BF16(a_hi, b_hi, acc[t]);
            acc[t] = WMMA_BF16(a_hi, b_lo, acc[t]);
            acc[t] = WMMA_BF16(a_lo, b_hi, acc[t]);
        }
        aptr += 512;
        bptr += 512;
    }

    // Epilogue: C/D layout -> VGPR r holds row r + 8*half, col = l16
#pragma unroll
    for (int t = 0; t < 8; ++t) {
        const int col = nbase + t * 16 + l16;
        const float bv = bias[col];
#pragma unroll
        for (int r = 0; r < 8; ++r) {
            const int row = m0 + r + half * 8;
            float v = acc[t][r] + bv;
            if (resid) v += resid[(size_t)row * N + col];
            C[(size_t)row * N + col] = v;
        }
    }
}

// ---------------------------------------------------------------------------
// Sequential reservoir scan. One block per (batch, reservoir): 16 blocks.
// 1024 threads = 256 k-lanes x 4 j-segments; W rows live in VGPRs (fully
// unrolled), state in an LDS double buffer. 2 barriers per step.
// ---------------------------------------------------------------------------
__global__ __launch_bounds__(1024)
void liquid_scan(const float* __restrict__ x,      // [MROWS, DDIM]
                 const float* __restrict__ prev,   // [BATCH, DDIM]
                 const float* __restrict__ w_res,  // [NRES, RSIZE, RSIZE]
                 float* __restrict__ states,       // [MROWS, DDIM] f32
                 float* __restrict__ final_state)  // [BATCH, DDIM]
{
    const int b   = blockIdx.x >> 2;
    const int r   = blockIdx.x & 3;
    const int tid = threadIdx.x;
    const int k   = tid & 255;
    const int seg = tid >> 8;                   // 0..3

    __shared__ float sbuf[2][RSIZE];
    __shared__ float psum[4][RSIZE];

    float w[64];
    {
        const float4* w4 =
            reinterpret_cast<const float4*>(w_res + ((size_t)r * RSIZE + k) * RSIZE + seg * 64);
#pragma unroll
        for (int q = 0; q < 16; ++q) {
            const float4 v = w4[q];
            w[4 * q + 0] = v.x; w[4 * q + 1] = v.y; w[4 * q + 2] = v.z; w[4 * q + 3] = v.w;
        }
    }

    float st = 0.f;
    if (seg == 0) {
        st = prev[(size_t)b * DDIM + r * RSIZE + k];
        sbuf[0][k] = st;
    }
    __syncthreads();

    int cur = 0;
    for (int t = 0; t < TLEN; ++t) {
        float p = 0.f;
#pragma unroll
        for (int j = 0; j < 64; ++j)
            p = fmaf(sbuf[cur][seg * 64 + j], w[j], p);
        psum[seg][k] = p;
        __syncthreads();

        if (seg == 0) {
            const size_t row = (size_t)b * TLEN + t;
            const float pre = psum[0][k] + psum[1][k] + psum[2][k] + psum[3][k]
                            + x[row * DDIM + r * RSIZE + k];
            const float ns = (1.0f - LEAKC) * st + LEAKC * tanhf(pre);
            st = ns;
            sbuf[cur ^ 1][k] = ns;
            states[row * DDIM + r * RSIZE + k] = ns;
            if (t == TLEN - 1)
                final_state[(size_t)b * DDIM + r * RSIZE + k] = ns;
        }
        __syncthreads();
        cur ^= 1;
    }
}

// ---------------------------------------------------------------------------
// RMS norm over H per row, scaled by ln_w (in place on y).
// ---------------------------------------------------------------------------
__global__ __launch_bounds__(256)
void liquid_rmsnorm(float* __restrict__ y, const float* __restrict__ lnw, int H) {
    __shared__ float red[8];
    float* yr = y + (size_t)blockIdx.x * H;
    const int tid = threadIdx.x;

    float s = 0.f;
    for (int i = tid; i < H; i += 256) { const float v = yr[i]; s = fmaf(v, v, s); }
#pragma unroll
    for (int off = 16; off > 0; off >>= 1) s += __shfl_down(s, off, 32);
    if ((tid & 31) == 0) red[tid >> 5] = s;
    __syncthreads();
    if (tid == 0) {
        float tot = 0.f;
#pragma unroll
        for (int i = 0; i < 8; ++i) tot += red[i];
        red[0] = rsqrtf(tot / (float)H + EPSC);
    }
    __syncthreads();
    const float f = red[0];
    for (int i = tid; i < H; i += 256) yr[i] = yr[i] * f * lnw[i];
}

// ---------------------------------------------------------------------------
// Launch. Workspace (dwords), peak 104 MB with aliasing:
//   [0      .. 16.8M)  hiddenP (64 MB)   -> later reused: states f32 (32 MB)
//                                           + statesP at dword 8.4M (32 MB)
//   [16.8M  .. 18.9M)  w_inP   (8 MB)    -> later reused: w_outP (8 MB)
//   [18.9M  .. 27.3M)  x_proj  (32 MB)
// ---------------------------------------------------------------------------
extern "C" void kernel_launch(void* const* d_in, const int* in_sizes, int n_in,
                              void* d_out, int out_size, void* d_ws, size_t ws_size,
                              hipStream_t stream) {
    const float* hidden = (const float*)d_in[0];   // [B,T,H]
    const float* prev   = (const float*)d_in[1];   // [B,D]
    const float* w_in   = (const float*)d_in[2];   // [H,D]
    const float* b_in   = (const float*)d_in[3];   // [D]
    const float* w_res  = (const float*)d_in[4];   // [R,S,S]
    const float* w_out  = (const float*)d_in[5];   // [D,H]
    const float* b_out  = (const float*)d_in[6];   // [H]
    const float* ln_w   = (const float*)d_in[7];   // [H]

    float* y           = (float*)d_out;                       // [B,T,H]
    float* final_state = y + (size_t)MROWS * HID;             // [B,D]

    unsigned* wsu = (unsigned*)d_ws;
    unsigned* hiddenP  = wsu;                                          // 16,777,216 dw
    unsigned* w_inP    = wsu + (size_t)MROWS * HID;                    //  2,097,152 dw
    float*    x_proj   = (float*)(w_inP + (size_t)HID * DDIM);         //  8,388,608 dw
    float*    states   = (float*)wsu;                                  // alias hiddenP
    unsigned* statesP  = wsu + (size_t)MROWS * DDIM;                   // alias hiddenP hi-half
    unsigned* w_outP   = w_inP;                                        // alias w_inP

    // 1) one-time bf16 hi/lo splits into fragment-native layout
    {
        const int totalA = MROWS * HID / 2;                  // 8,388,608
        pack_split_rows<<<dim3(totalA / 256), dim3(256), 0, stream>>>(
            hidden, hiddenP, HID, 6 /* log2(2048/32) */, totalA);
        const int totalB = HID * DDIM / 2;                   // 1,048,576
        pack_split_cols<<<dim3(totalB / 256), dim3(256), 0, stream>>>(
            w_in, w_inP, DDIM, 6, totalB);
    }

    // 2) x = hidden @ w_in + b_in         (M=8192, N=1024, K=2048)
    liquid_gemm_wmma<<<dim3(DDIM / 512, MROWS / 16), dim3(128), 0, stream>>>(
        hiddenP, w_inP, b_in, nullptr, x_proj, MROWS, DDIM, HID / 32);

    // 3) sequential reservoir scan (writes f32 states into reused region)
    liquid_scan<<<dim3(BATCH * NRES), dim3(1024), 0, stream>>>(
        x_proj, prev, w_res, states, final_state);

    // 4) repack states + w_out for the output GEMM
    {
        const int totalS = MROWS * DDIM / 2;                 // 4,194,304
        pack_split_rows<<<dim3(totalS / 256), dim3(256), 0, stream>>>(
            states, statesP, DDIM, 5 /* log2(1024/32) */, totalS);
        const int totalW = DDIM * HID / 2;                   // 1,048,576
        pack_split_cols<<<dim3(totalW / 256), dim3(256), 0, stream>>>(
            w_out, w_outP, HID, 5, totalW);
    }

    // 5) y_pre = hidden + states @ w_out + b_out   (M=8192, N=2048, K=1024)
    liquid_gemm_wmma<<<dim3(HID / 512, MROWS / 16), dim3(128), 0, stream>>>(
        statesP, w_outP, b_out, hidden, y, MROWS, HID, DDIM / 32);

    // 6) RMS norm * ln_w (in place on y)
    liquid_rmsnorm<<<dim3(MROWS), dim3(256), 0, stream>>>(y, ln_w, HID);
}
